// EncoderLayerModule_61993557950768
// MI455X (gfx1250) — matrix operand order repaired and checked by reading, not compiled, not run on hardware
//
#include <hip/hip_runtime.h>
#include <hip/hip_bf16.h>
#include <math.h>

typedef __attribute__((ext_vector_type(16))) _Float16 v16h;
typedef __attribute__((ext_vector_type(8)))  _Float16 v8h;
typedef __attribute__((ext_vector_type(8)))  float    v8f;

#define BB 16
#define CC 32
#define LL 512
#define DD 512
#define TT 65
#define NHEAD 8
#define DH 64
#define RR (BB*CC)        /* 512  */
#define MM (RR*TT)        /* 33280 */

// ---------------------------------------------------------------- elementwise
__global__ void f32_to_f16_kernel(const float* __restrict__ src,
                                  _Float16* __restrict__ dst, int n) {
  int i = blockIdx.x * 256 + threadIdx.x;
  if (i < n) dst[i] = (_Float16)src[i];
}

__global__ void magphase_kernel(const float* __restrict__ xr, const float* __restrict__ xi,
                                float* __restrict__ mag, float* __restrict__ ph) {
  int i = blockIdx.x * 256 + threadIdx.x;
  if (i >= BB * CC * LL) return;
  float a = xr[i], b = xi[i];
  mag[i] = sqrtf(a * a + b * b);
  ph[i]  = atan2f(b, a);
}

// gm[l] = sum over B*C rows of mag^2  (deterministic tree reduction)
__global__ void energy_kernel(const float* __restrict__ mag, float* __restrict__ gm) {
  __shared__ float s[256];
  int l = blockIdx.x, tid = threadIdx.x;
  float m0 = mag[(size_t)tid * LL + l];
  float m1 = mag[(size_t)(tid + 256) * LL + l];
  s[tid] = m0 * m0 + m1 * m1;
  __syncthreads();
  for (int st = 128; st > 0; st >>= 1) {
    if (tid < st) s[tid] += s[tid + st];
    __syncthreads();
  }
  if (tid == 0) gm[l] = s[0];
}

// global top-32 + 3 bands of top-11 (stable argsort tie-break == strict '>')
__global__ void topk_kernel(const float* __restrict__ gm, int* __restrict__ idx) {
  if (threadIdx.x != 0 || blockIdx.x != 0) return;
  bool used[LL];
  for (int i = 0; i < LL; ++i) used[i] = false;
  int pos = 0;
  for (int t = 0; t < 32; ++t) {
    float best = -3.4e38f; int bi = 0;
    for (int l = 0; l < LL; ++l)
      if (!used[l] && gm[l] > best) { best = gm[l]; bi = l; }
    used[bi] = true;
    idx[pos++] = bi;
  }
  const int band = LL / 3;               // 170
  for (int bd = 0; bd < 3; ++bd) {
    int s = bd * band;
    int e = (bd == 2) ? LL : (bd + 1) * band;
    for (int i = s; i < e; ++i) used[i] = false;   // bands selected independently
    for (int t = 0; t < 11; ++t) {
      float best = -3.4e38f; int bi = s;
      for (int l = s; l < e; ++l)
        if (!used[l] && gm[l] > best) { best = gm[l]; bi = l; }
      used[bi] = true;
      idx[pos++] = bi;
    }
  }
}

__global__ void gather_kernel(const float* __restrict__ mag, const float* __restrict__ ph,
                              const int* __restrict__ idx,
                              float* __restrict__ km, float* __restrict__ kp,
                              float* __restrict__ kr, float* __restrict__ ki) {
  int e = blockIdx.x * 256 + threadIdx.x;
  if (e >= RR * TT) return;
  int rrow = e / TT, t = e % TT;
  int li = idx[t];
  float m = mag[(size_t)rrow * LL + li];
  float p = ph[(size_t)rrow * LL + li];
  km[e] = m; kp[e] = p;
  kr[e] = m * cosf(p);
  ki[e] = m * sinf(p);
}

// ---------------------------------------------------------------- gumbel mask
__global__ void dist_kernel(const float* __restrict__ km, const float* __restrict__ kp,
                            const float* __restrict__ Amat,
                            const float* __restrict__ wa_p, const float* __restrict__ wp_p,
                            float* __restrict__ expd) {
  int e = blockIdx.x * 256 + threadIdx.x;
  if (e >= BB * TT * TT) return;
  int b = e / (TT * TT); int ij = e % (TT * TT);
  int i = ij / TT, j = ij % TT;
  float wa = wa_p[0], wp = wp_p[0];
  float Aj = Amat[j * TT + j];
  float dist = 0.f;
  for (int c = 0; c < CC; ++c) {
    size_t base = (size_t)(b * CC + c) * TT;
    float f = wa * (km[base + i] - km[base + j]) + wp * (kp[base + i] - kp[base + j]);
    float t2 = Aj * f;
    dist += t2 * t2;
  }
  expd[e] = (i == j) ? 0.f : 1.f / (dist + 1e-10f);
}

__global__ void rowmax_kernel(const float* __restrict__ expd, float* __restrict__ rmax) {
  int e = blockIdx.x * 256 + threadIdx.x;
  if (e >= BB * TT) return;
  int b = e / TT, i = e % TT;
  const float* row = expd + (size_t)b * TT * TT + (size_t)i * TT;
  float mx = 0.f;
  for (int j = 0; j < TT; ++j) mx = fmaxf(mx, row[j]);
  rmax[e] = mx;
}

__global__ void mask_kernel(const float* __restrict__ expd, const float* __restrict__ rmax,
                            const float* __restrict__ noise, float* __restrict__ mask) {
  int ij = blockIdx.x * 256 + threadIdx.x;
  if (ij >= TT * TT) return;
  int i = ij / TT, j = ij % TT;
  float acc = 0.f;
  for (int b = 0; b < BB; ++b) {
    float p = (i == j) ? 0.99f
                       : 0.99f * expd[(size_t)b * TT * TT + ij] / rmax[b * TT + i];
    float lg = logf(p / (1.f - p));
    size_t ni = ((size_t)(b * TT + i) * TT + j) * 2;
    float y0 = lg + noise[ni];
    float y1 = -lg + noise[ni + 1];
    acc += (y0 >= y1) ? 1.f : 0.f;       // argmax tie -> index 0
  }
  mask[ij] = acc * (1.f / BB);
}

// ---------------------------------------------------------------- rel (h init)
__global__ void rel_kernel(const float* __restrict__ xr, const float* __restrict__ xi,
                           const float* __restrict__ kr, const float* __restrict__ ki,
                           float* __restrict__ H, _Float16* __restrict__ HH) {
  size_t e = (size_t)blockIdx.x * 256 + threadIdx.x;
  if (e >= (size_t)MM * DD) return;
  int l = (int)(e % DD);
  size_t mrow = e / DD;
  int t = (int)(mrow % TT);
  int rrow = (int)(mrow / TT);
  size_t xoff = (size_t)rrow * LL + l;
  size_t koff = (size_t)rrow * TT + t;
  float krv = kr[koff], kiv = ki[koff];
  float xrv = xr[xoff], xiv = xi[xoff];
  float a   = krv * xrv - kiv * xiv;
  float bb2 = krv * xiv + kiv * xrv;
  float ea = expf(-a);
  float wr = 1.f + ea * cosf(bb2);
  float wi = -ea * sinf(bb2);
  float v = wr / (wr * wr + wi * wi);
  H[e] = v;
  HH[e] = (_Float16)v;
}

// ---------------------------------------------------------------- WMMA GEMM
// out[m,n] = sum_k A[m,k] * W[n,k] + bias[n]   (x @ W^T + b), M=MM, N=K=512.
// Block tile 128Mx128N staged in LDS; 8 waves in 4x2 grid; each wave 32Mx64N
// (8 accumulators), K chunked by 64 (2 WMMA K-steps / barrier).
#define KCH  64
#define ASTR 72   /* padded LDS row stride in f16: 2-way-optimal banks for b128 */

__device__ __forceinline__ float gelu_exact(float x) {
  return 0.5f * x * (1.f + erff(x * 0.70710678118654752f));
}

template<int EPI>   // 0: store f16, 1: GELU + f16, 2: store f32
__global__ void gemm_wmma_kernel(const _Float16* __restrict__ Ag,
                                 const _Float16* __restrict__ Wg,
                                 const float* __restrict__ bias,
                                 _Float16* __restrict__ out16,
                                 float* __restrict__ out32) {
  __shared__ _Float16 sA[128 * ASTR];
  __shared__ _Float16 sB[128 * ASTR];
  const int bx   = blockIdx.x;
  const int nb   = bx & 3;           // N block 0..3
  const int mb   = bx >> 2;          // M block 0..259
  const int m0   = mb * 128;
  const int n0   = nb * 128;
  const int tid  = threadIdx.x;
  const int wv   = tid >> 5;
  const int lane = tid & 31;
  const int half = lane >> 4;
  const int r    = lane & 15;
  const int wm   = (wv >> 1) * 32;   // wave M offset in tile (0,32,64,96)
  const int wn   = (wv & 1) * 64;    // wave N offset in tile (0,64)

  v8f acc[2][4] = {};

  for (int kc = 0; kc < DD; kc += KCH) {
    if (kc) __syncthreads();         // previous chunk fully consumed
    // ---- stage A[128][64] and B[128][64] (B row n = W[n, k..]) into LDS
    for (int e = tid; e < 128 * (KCH / 8); e += 256) {   // 1024 chunks of 8 f16
      int row = e >> 3, c8 = (e & 7) * 8;
      v8h av = *(const v8h*)(Ag + (size_t)(m0 + row) * DD + kc + c8);
      v8h bv = *(const v8h*)(Wg + (size_t)(n0 + row) * DD + kc + c8);
      *(v8h*)(sA + row * ASTR + c8) = av;
      *(v8h*)(sB + row * ASTR + c8) = bv;
      if (kc + KCH < DD) {
        __builtin_prefetch(Ag + (size_t)(m0 + row) * DD + kc + KCH + c8, 0, 1);
        __builtin_prefetch(Wg + (size_t)(n0 + row) * DD + kc + KCH + c8, 0, 1);
      }
    }
    __syncthreads();
    // ---- compute: 2 K-steps x 8 WMMA from LDS fragments
#pragma unroll
    for (int kb = 0; kb < KCH; kb += 32) {
      union U { v16h v; v8h h[2]; };
      U a0, a1, b0, b1, b2, b3;
      const _Float16* ap = sA + (wm + r) * ASTR + kb + half * 16;
      a0.h[0] = *(const v8h*)(ap);
      a0.h[1] = *(const v8h*)(ap + 8);
      a1.h[0] = *(const v8h*)(ap + 16 * ASTR);
      a1.h[1] = *(const v8h*)(ap + 16 * ASTR + 8);
      const _Float16* bp = sB + (wn + r) * ASTR + kb + half * 16;
      b0.h[0] = *(const v8h*)(bp);
      b0.h[1] = *(const v8h*)(bp + 8);
      b1.h[0] = *(const v8h*)(bp + 16 * ASTR);
      b1.h[1] = *(const v8h*)(bp + 16 * ASTR + 8);
      b2.h[0] = *(const v8h*)(bp + 32 * ASTR);
      b2.h[1] = *(const v8h*)(bp + 32 * ASTR + 8);
      b3.h[0] = *(const v8h*)(bp + 48 * ASTR);
      b3.h[1] = *(const v8h*)(bp + 48 * ASTR + 8);
      acc[0][0] = __builtin_amdgcn_wmma_f32_16x16x32_f16(false, a0.v, false, b0.v, (short)0, acc[0][0], false, false);
      acc[0][1] = __builtin_amdgcn_wmma_f32_16x16x32_f16(false, a0.v, false, b1.v, (short)0, acc[0][1], false, false);
      acc[0][2] = __builtin_amdgcn_wmma_f32_16x16x32_f16(false, a0.v, false, b2.v, (short)0, acc[0][2], false, false);
      acc[0][3] = __builtin_amdgcn_wmma_f32_16x16x32_f16(false, a0.v, false, b3.v, (short)0, acc[0][3], false, false);
      acc[1][0] = __builtin_amdgcn_wmma_f32_16x16x32_f16(false, a1.v, false, b0.v, (short)0, acc[1][0], false, false);
      acc[1][1] = __builtin_amdgcn_wmma_f32_16x16x32_f16(false, a1.v, false, b1.v, (short)0, acc[1][1], false, false);
      acc[1][2] = __builtin_amdgcn_wmma_f32_16x16x32_f16(false, a1.v, false, b2.v, (short)0, acc[1][2], false, false);
      acc[1][3] = __builtin_amdgcn_wmma_f32_16x16x32_f16(false, a1.v, false, b3.v, (short)0, acc[1][3], false, false);
    }
  }

  // ---- epilogue (D layout: lanes>=16 own rows +8; col = lane&15)
#pragma unroll
  for (int ms = 0; ms < 2; ++ms) {
#pragma unroll
    for (int nt = 0; nt < 4; ++nt) {
      int n = n0 + wn + nt * 16 + r;
      float bvs = bias[n];
#pragma unroll
      for (int v = 0; v < 8; ++v) {
        float val = acc[ms][nt][v] + bvs;
        if (EPI == 1) val = gelu_exact(val);
        size_t offo = (size_t)(m0 + wm + ms * 16 + half * 8 + v) * DD + n;
        if (EPI == 2) out32[offo] = val;
        else          out16[offo] = (_Float16)val;
      }
    }
  }
}

// ---------------------------------------------------------------- attention
__global__ void attn_kernel(const _Float16* __restrict__ Q, const _Float16* __restrict__ K,
                            const _Float16* __restrict__ V, const float* __restrict__ mask,
                            _Float16* __restrict__ O) {
  __shared__ _Float16 sQ[TT * DH], sK[TT * DH], sV[TT * DH];
  __shared__ float sS[TT * TT];
  int bh = blockIdx.x;
  int bn = bh >> 3, hd = bh & 7;
  size_t rowbase = (size_t)bn * TT;
  int ch = hd * DH;
  int tid = threadIdx.x;
  for (int e = tid; e < TT * DH; e += 256) {
    int i = e / DH, d2 = e % DH;
    size_t src = (rowbase + i) * DD + ch + d2;
    sQ[e] = Q[src]; sK[e] = K[src]; sV[e] = V[src];
  }
  __syncthreads();
  for (int e = tid; e < TT * TT; e += 256) {
    int i = e / TT, j = e % TT;
    float acc = 0.f;
#pragma unroll 8
    for (int d2 = 0; d2 < DH; ++d2)
      acc += (float)sQ[i * DH + d2] * (float)sK[j * DH + d2];
    float m = mask[e];
    sS[e] = acc * m + (1.f - m) * (-1e9f);
  }
  __syncthreads();
  if (tid < TT) {                       // row softmax of (masked scores / 8)
    float mx = -3.4e38f;
    for (int j = 0; j < TT; ++j) {
      float v = sS[tid * TT + j] * 0.125f;
      sS[tid * TT + j] = v;
      if (v > mx) mx = v;
    }
    float s = 0.f;
    for (int j = 0; j < TT; ++j) {
      float e = expf(sS[tid * TT + j] - mx);
      sS[tid * TT + j] = e; s += e;
    }
    float inv = 1.f / s;
    for (int j = 0; j < TT; ++j) sS[tid * TT + j] *= inv;
  }
  __syncthreads();
  for (int e = tid; e < TT * DH; e += 256) {
    int i = e / DH, d2 = e % DH;
    float acc = 0.f;
    for (int j = 0; j < TT; ++j) acc += sS[i * TT + j] * (float)sV[j * DH + d2];
    O[(rowbase + i) * DD + ch + d2] = (_Float16)acc;
  }
}

// ---------------------------------------------------------------- layernorm
__global__ void ln_kernel(const float* __restrict__ x, const float* __restrict__ res,
                          const float* __restrict__ g, const float* __restrict__ b,
                          float* __restrict__ y, _Float16* __restrict__ y16) {
  __shared__ float s1[256], s2[256];
  int row = blockIdx.x, tid = threadIdx.x;
  size_t base = (size_t)row * DD;
  float v0 = x[base + tid];
  float v1 = x[base + tid + 256];
  if (res) { v0 += res[base + tid]; v1 += res[base + tid + 256]; }
  s1[tid] = v0 + v1;
  s2[tid] = v0 * v0 + v1 * v1;
  __syncthreads();
  for (int st = 128; st > 0; st >>= 1) {
    if (tid < st) { s1[tid] += s1[tid + st]; s2[tid] += s2[tid + st]; }
    __syncthreads();
  }
  float mean = s1[0] * (1.f / DD);
  float var  = s2[0] * (1.f / DD) - mean * mean;
  float inv = rsqrtf(var + 1e-5f);
  float o0 = (v0 - mean) * inv * g[tid] + b[tid];
  float o1 = (v1 - mean) * inv * g[tid + 256] + b[tid + 256];
  y[base + tid] = o0; y[base + tid + 256] = o1;
  if (y16) { y16[base + tid] = (_Float16)o0; y16[base + tid + 256] = (_Float16)o1; }
}

// ---------------------------------------------------------------- output
__global__ void output_kernel(const float* __restrict__ H, const float* __restrict__ km,
                              const float* __restrict__ ph, float* __restrict__ out) {
  __shared__ float sw[TT];
  int rrow = blockIdx.x, tid = threadIdx.x;
  if (tid == 0) {
    float mx = -3.4e38f;
    for (int t = 0; t < TT; ++t) mx = fmaxf(mx, km[rrow * TT + t]);
    float s = 0.f;
    for (int t = 0; t < TT; ++t) { float e = expf(km[rrow * TT + t] - mx); sw[t] = e; s += e; }
    float inv = 1.f / s;
    for (int t = 0; t < TT; ++t) sw[t] *= inv;
  }
  __syncthreads();
  for (int d = tid; d < DD; d += 256) {
    float fs = 0.f;
    for (int t = 0; t < TT; ++t)
      fs += H[((size_t)rrow * TT + t) * DD + d] * sw[t];
    float pv = ph[(size_t)rrow * LL + d];
    size_t o = ((size_t)rrow * LL + d) * 2;
    out[o]     = fs * cosf(pv);
    out[o + 1] = fs * sinf(pv);
  }
}

// ---------------------------------------------------------------- launch
extern "C" void kernel_launch(void* const* d_in, const int* in_sizes, int n_in,
                              void* d_out, int out_size, void* d_ws, size_t ws_size,
                              hipStream_t stream) {
  (void)in_sizes; (void)n_in; (void)out_size; (void)ws_size;
  const float* xr    = (const float*)d_in[0];
  const float* xi    = (const float*)d_in[1];
  const float* noise = (const float*)d_in[2];
  const float* Amat  = (const float*)d_in[3];
  const float* wa    = (const float*)d_in[4];
  const float* wp    = (const float*)d_in[5];
  const float* Wsrc[6] = {(const float*)d_in[6],  (const float*)d_in[8],
                          (const float*)d_in[10], (const float*)d_in[12],
                          (const float*)d_in[14], (const float*)d_in[16]};
  const float* Bsrc[6] = {(const float*)d_in[7],  (const float*)d_in[9],
                          (const float*)d_in[11], (const float*)d_in[13],
                          (const float*)d_in[15], (const float*)d_in[17]};
  const float* g1 = (const float*)d_in[18];
  const float* b1 = (const float*)d_in[19];
  const float* g2 = (const float*)d_in[20];
  const float* b2 = (const float*)d_in[21];
  const float* gf = (const float*)d_in[22];
  const float* bf = (const float*)d_in[23];
  float* out = (float*)d_out;

  char* ws = (char*)d_ws;
  size_t cur = 0;
  auto alloc = [&](size_t bytes) {
    size_t r = cur; cur = (cur + bytes + 255) & ~(size_t)255; return r;
  };
  const size_t wsz  = (size_t)DD * DD;                  // 262144 elems per matrix
  const size_t hsz  = (size_t)MM * DD;                  // 17,039,360 elems
  _Float16* W16  = (_Float16*)(ws + alloc(12 * wsz * 2));
  float*    MAG  = (float*)(ws + alloc((size_t)BB * CC * LL * 4));
  float*    PH   = (float*)(ws + alloc((size_t)BB * CC * LL * 4));
  float*    GM   = (float*)(ws + alloc(LL * 4));
  int*      IDX  = (int*)  (ws + alloc(TT * 4));
  float*    KEYM = (float*)(ws + alloc((size_t)RR * TT * 4));
  float*    KEYP = (float*)(ws + alloc((size_t)RR * TT * 4));
  float*    KR   = (float*)(ws + alloc((size_t)RR * TT * 4));
  float*    KI   = (float*)(ws + alloc((size_t)RR * TT * 4));
  float*    EXPD = (float*)(ws + alloc((size_t)BB * TT * TT * 4));
  float*    RMAX = (float*)(ws + alloc((size_t)BB * TT * 4));
  float*    MASK = (float*)(ws + alloc((size_t)TT * TT * 4));
  float*    H    = (float*)(ws + alloc(hsz * 4));
  _Float16* HH   = (_Float16*)(ws + alloc(hsz * 2));
  _Float16* Q16  = (_Float16*)(ws + alloc(hsz * 2));   // Q16,K16 contiguous:
  _Float16* K16  = (_Float16*)(ws + alloc(hsz * 2));   //   TBUF aliases both (f32)
  _Float16* V16  = (_Float16*)(ws + alloc(hsz * 2));   //   Y16 aliases V16
  _Float16* O16  = (_Float16*)(ws + alloc(hsz * 2));
  float*    TBUF = (float*)Q16;      // 2 x f16 region == 1 x f32 region, dead Q/K
  _Float16* Y16  = V16;              // FFN intermediate over dead V
  (void)K16;

  // ---- weight conversion to f16
  for (int w = 0; w < 6; ++w)
    f32_to_f16_kernel<<<(int)((2 * wsz + 255) / 256), 256, 0, stream>>>(
        Wsrc[w], W16 + (size_t)w * 2 * wsz, (int)(2 * wsz));

  // ---- spectrum prep
  magphase_kernel<<<(BB * CC * LL) / 256, 256, 0, stream>>>(xr, xi, MAG, PH);
  energy_kernel<<<LL, 256, 0, stream>>>(MAG, GM);
  topk_kernel<<<1, 32, 0, stream>>>(GM, IDX);
  gather_kernel<<<(RR * TT + 255) / 256, 256, 0, stream>>>(MAG, PH, IDX, KEYM, KEYP, KR, KI);

  // ---- gumbel relation mask
  dist_kernel<<<(BB * TT * TT + 255) / 256, 256, 0, stream>>>(KEYM, KEYP, Amat, wa, wp, EXPD);
  rowmax_kernel<<<(BB * TT + 255) / 256, 256, 0, stream>>>(EXPD, RMAX);
  mask_kernel<<<(TT * TT + 255) / 256, 256, 0, stream>>>(EXPD, RMAX, noise, MASK);

  // ---- h init (complex sigmoid relationship)
  rel_kernel<<<(int)(hsz / 256), 256, 0, stream>>>(xr, xi, KR, KI, H, HH);

  // ---- 2 encoder layers
  const int GB = (MM / 128) * (DD / 128);     // 260 * 4 = 1040 GEMM blocks
  for (int l = 0; l < 2; ++l) {
    const _Float16* Wq16  = W16 + (size_t)(0 * 2 + l) * wsz;
    const _Float16* Wk16  = W16 + (size_t)(1 * 2 + l) * wsz;
    const _Float16* Wv16  = W16 + (size_t)(2 * 2 + l) * wsz;
    const _Float16* Wo16  = W16 + (size_t)(3 * 2 + l) * wsz;
    const _Float16* Wc116 = W16 + (size_t)(4 * 2 + l) * wsz;
    const _Float16* Wc216 = W16 + (size_t)(5 * 2 + l) * wsz;

    gemm_wmma_kernel<0><<<GB, 256, 0, stream>>>(HH, Wq16, Bsrc[0] + l * DD, Q16, nullptr);
    gemm_wmma_kernel<0><<<GB, 256, 0, stream>>>(HH, Wk16, Bsrc[1] + l * DD, K16, nullptr);
    gemm_wmma_kernel<0><<<GB, 256, 0, stream>>>(HH, Wv16, Bsrc[2] + l * DD, V16, nullptr);
    attn_kernel<<<RR * NHEAD, 256, 0, stream>>>(Q16, K16, V16, MASK, O16);
    gemm_wmma_kernel<2><<<GB, 256, 0, stream>>>(O16, Wo16, Bsrc[3] + l * DD, nullptr, TBUF);
    ln_kernel<<<MM, 256, 0, stream>>>(H, TBUF, g1 + l * DD, b1 + l * DD, H, HH);
    gemm_wmma_kernel<1><<<GB, 256, 0, stream>>>(HH, Wc116, Bsrc[4] + l * DD, Y16, nullptr);
    gemm_wmma_kernel<2><<<GB, 256, 0, stream>>>(Y16, Wc216, Bsrc[5] + l * DD, nullptr, TBUF);
    ln_kernel<<<MM, 256, 0, stream>>>(H, TBUF, g2 + l * DD, b2 + l * DD, H, HH);
  }

  // ---- final LN + weighted spectrum + polar output
  ln_kernel<<<MM, 256, 0, stream>>>(H, nullptr, gf, bf, H, nullptr);
  output_kernel<<<RR, 256, 0, stream>>>(H, KEYM, PH, out);
}
